// S4Encoder_55362128445893
// MI455X (gfx1250) — matrix operand order repaired and checked by reading, not compiled, not run on hardware
//
#include <hip/hip_runtime.h>
#include <math.h>

// ---------------- problem constants ----------------
constexpr int kH    = 512;           // hidden
constexpr int kNL   = 4;             // layers
constexpr int kN2   = 32;            // d_state/2 (== wave32!)
constexpr int kB    = 8;
constexpr int kL    = 4096;
constexpr int kM    = kB * kL;       // 32768 tokens
constexpr int kT    = 64;            // SSM chunk length
constexpr int kNC   = kL / kT;       // 64 chunks

typedef unsigned short bf16_t;
typedef __attribute__((ext_vector_type(16))) __bf16 v16bf;
typedef __attribute__((ext_vector_type(8)))  float  v8f;

union BF16x16 { v16bf v; bf16_t u[16]; uint4 q[2]; };

// ---------------- CDNA5 async global->LDS (guarded) ----------------
#if defined(__has_builtin)
#if __has_builtin(__builtin_amdgcn_global_load_async_to_lds_b32) && \
    __has_builtin(__builtin_amdgcn_global_load_async_to_lds_b128)
#define ASYNC_LDS 1
#endif
#endif

#if defined(ASYNC_LDS)
typedef int vsi4 __attribute__((vector_size(16)));
typedef __attribute__((address_space(1))) int  gint_t;
typedef __attribute__((address_space(3))) int  lint_t;
typedef __attribute__((address_space(1))) vsi4 gvsi4_t;
typedef __attribute__((address_space(3))) vsi4 lvsi4_t;
__device__ inline void async_copy_b32(const bf16_t* g, bf16_t* l) {
  __builtin_amdgcn_global_load_async_to_lds_b32((gint_t*)g, (lint_t*)l, 0, 0);
}
__device__ inline void async_copy_b128(const bf16_t* g, bf16_t* l) {
  __builtin_amdgcn_global_load_async_to_lds_b128((gvsi4_t*)g, (lvsi4_t*)l, 0, 0);
}
__device__ inline void wait_async0() {
#if __has_builtin(__builtin_amdgcn_s_wait_asynccnt)
  __builtin_amdgcn_s_wait_asynccnt(0);
#else
  asm volatile("s_wait_asynccnt 0x0" ::: "memory");
#endif
}
#endif

__device__ inline bf16_t f2bf(float f) {
  union { float f; unsigned u; } x; x.f = f;
  unsigned r = x.u + 0x7fffu + ((x.u >> 16) & 1u);   // round-to-nearest-even
  return (bf16_t)(r >> 16);
}

__device__ inline v8f vzero8() {
  v8f z;
#pragma unroll
  for (int i = 0; i < 8; ++i) z[i] = 0.f;
  return z;
}

__device__ inline v8f wmma_bf16(const BF16x16& a, const BF16x16& b, v8f c) {
  return __builtin_amdgcn_wmma_f32_16x16x32_bf16(false, a.v, false, b.v,
                                                 (short)0, c, false, false);
}

// A-fragment (16x32, 16-bit): lane row = lane&15, kb = 8*(lane>>4);
// element e -> K = kb+e (e<8) else 16+kb+(e-8)  => two 16B loads with a gap.
__device__ inline BF16x16 load_gap16(const bf16_t* p) {
  BF16x16 f;
  f.q[0] = *(const uint4*)(p);
  f.q[1] = *(const uint4*)(p + 16);
  return f;
}
// 16 contiguous bf16 (used when reading B as a transposed row: K contiguous).
__device__ inline BF16x16 load_seq16(const bf16_t* p) {
  BF16x16 f;
  f.q[0] = *(const uint4*)(p);
  f.q[1] = *(const uint4*)(p + 8);
  return f;
}

// ---------------- transpose x[b][l][h] -> Xt bf16 [h][b*L+l] ----------------
__global__ void k_tr(const float* __restrict__ x, bf16_t* __restrict__ Xt) {
  __shared__ float tile[32][33];
  int b = blockIdx.x, l0 = blockIdx.y * 32, h0 = blockIdx.z * 32;
  int tx = threadIdx.x, ty = threadIdx.y;
  for (int r = ty; r < 32; r += 8)
    tile[r][tx] = x[(size_t)(b * kL + l0 + r) * kH + h0 + tx];
  __syncthreads();
  for (int r = ty; r < 32; r += 8)
    Xt[(size_t)(h0 + r) * kM + b * kL + l0 + tx] = f2bf(tile[tx][r]);
}

// ---------------- convert all layer weights W -> bf16 ----------------
__global__ void k_wcvt(const float* __restrict__ W, bf16_t* __restrict__ Wb, int n) {
  int i = blockIdx.x * 256 + threadIdx.x;
  if (i < n) Wb[i] = f2bf(W[i]);
}

// ---------------- per-layer S4D tables: Q, Mk(Toeplitz+D), P, lambda^T -------
__global__ void k_gen(int lay,
                      const float* __restrict__ log_dt, const float* __restrict__ A_imag,
                      const float* __restrict__ log_A_real, const float* __restrict__ Cp,
                      const float* __restrict__ Dp,
                      bf16_t* __restrict__ Qb, bf16_t* __restrict__ Mkb,
                      bf16_t* __restrict__ Pb, float* __restrict__ LamT) {
  int h = blockIdx.x, n = threadIdx.x;        // one wave32, lane == state index
  __shared__ float Ksh[kT];
  int hi  = lay * kH + h;
  int idx = hi * kN2 + n;
  float dt  = expf(log_dt[hi]);
  float Ar  = -expf(log_A_real[idx]);
  float Ai  = A_imag[idx];
  float dAr = Ar * dt, dAi = Ai * dt;
  float er  = expf(dAr);
  float lr  = er * cosf(dAi), li = er * sinf(dAi);     // lambda = exp(dA)
  float Cr  = Cp[idx * 2 + 0], Ci = Cp[idx * 2 + 1];
  float nr  = lr - 1.f, ni = li;
  float den = Ar * Ar + Ai * Ai;
  float tr  = (nr * Ar + ni * Ai) / den;
  float ti  = (ni * Ar - nr * Ai) / den;               // (lambda-1)/A
  float cmr = Cr * tr - Ci * ti, cmi = Cr * ti + Ci * tr;

  bf16_t* q  = Qb + (size_t)h * (kT * kT);
  bf16_t* pp = Pb + (size_t)h * (kT * kT);
  float pr = 1.f, pi = 0.f;                            // lambda^0
  for (int p = 0; p < kT; ++p) {
    float part = cmr * pr - cmi * pi;
#pragma unroll
    for (int off = 16; off > 0; off >>= 1) part += __shfl_xor(part, off, 32);
    if (n == 0) Ksh[p] = 2.f * part;                   // K[p] = 2Re(sum c*lam^p)
    q[(2 * n + 0) * kT + (kT - 1 - p)] = f2bf(pr);     // s_loc = sum lam^{63-j} u_j
    q[(2 * n + 1) * kT + (kT - 1 - p)] = f2bf(pi);
    float t0 = pr * lr - pi * li; pi = pr * li + pi * lr; pr = t0;  // lambda^{p+1}
    pp[p * kT + 2 * n + 0] = f2bf( 2.f * (cmr * pr - cmi * pi));   // * s_re
    pp[p * kT + 2 * n + 1] = f2bf(-2.f * (cmi * pr + cmr * pi));   // * s_im
  }
  LamT[(h * kN2 + n) * 2 + 0] = pr;                    // lambda^64
  LamT[(h * kN2 + n) * 2 + 1] = pi;
  __syncthreads();
  float Dh = Dp[hi];
  bf16_t* mk = Mkb + (size_t)h * (kT * kT);
  for (int t = 0; t < kT; ++t)
    for (int j = n; j < kT; j += 32) {
      float v = 0.f;
      if (j < t)       v = Ksh[t - j];
      else if (j == t) v = Ksh[0] + Dh;                // fold D*u into diagonal
      mk[t * kT + j] = f2bf(v);
    }
}

// ---------------- GEMM1: S_locT[m'][64] = U^T (64) x Q^T  (per h) -----------
__global__ __launch_bounds__(256) void k_gemm1(const bf16_t* __restrict__ Xt,
                                               const bf16_t* __restrict__ Qb,
                                               float* __restrict__ SlocT) {
  int h = blockIdx.x;
  int tid = threadIdx.x, lane = tid & 31, w = tid >> 5;
  __shared__ __align__(16) bf16_t lds_q[kT * kT];
#if defined(ASYNC_LDS)
  for (int i = tid * 8; i < kT * kT; i += 256 * 8)
    async_copy_b128(&Qb[(size_t)h * (kT * kT) + i], &lds_q[i]);
  wait_async0();
#else
  for (int i = tid * 8; i < kT * kT; i += 256 * 8)
    *(uint4*)&lds_q[i] = *(const uint4*)&Qb[(size_t)h * (kT * kT) + i];
#endif
  __syncthreads();

  int m0 = blockIdx.y * 128 + w * 16;                  // m' row-tile of OUT^T
  int mp = m0 + (lane & 15);
  int kb = (lane >> 4) << 3, cb = (lane >> 4) << 4;
  const bf16_t* ub = Xt + (size_t)h * kM + (mp >> 6) * kL + (mp & 63) * kT;
  v8f acc[4];
#pragma unroll
  for (int i = 0; i < 4; ++i) acc[i] = vzero8();
#pragma unroll
  for (int kk = 0; kk < kT; kk += 32) {
    BF16x16 af = load_seq16(ub + kk + cb);
#pragma unroll
    for (int ct = 0; ct < 4; ++ct) {
      BF16x16 bfr = load_seq16(&lds_q[(ct * 16 + (lane & 15)) * kT + kk + cb]);
      acc[ct] = wmma_bf16(af, bfr, acc[ct]);
    }
  }
  int rb = m0 + 8 * (lane >> 4), cl = lane & 15;
#pragma unroll
  for (int ct = 0; ct < 4; ++ct)
#pragma unroll
    for (int r = 0; r < 8; ++r)
      SlocT[((size_t)h * 512 + rb + r) * kT + ct * 16 + cl] = acc[ct][r];
}

// ---------------- chunk scan: s_pre[c+1] = lam^64 * s_pre[c] + s_loc[c] ------
__global__ void k_cscan(const float* __restrict__ LamT, const float* __restrict__ SlocT,
                        bf16_t* __restrict__ SpreT) {
  int h = blockIdx.x >> 3, b = blockIdx.x & 7, n = threadIdx.x;
  float lr = LamT[(h * kN2 + n) * 2 + 0], li = LamT[(h * kN2 + n) * 2 + 1];
  const float* sl = SlocT + ((size_t)h * 512 + b * kNC) * kT + 2 * n;
  bf16_t*      sp = SpreT + ((size_t)h * 512 + b * kNC) * kT + 2 * n;
  float sr = 0.f, si = 0.f;
  for (int c = 0; c < kNC; ++c) {
    unsigned pk = ((unsigned)f2bf(si) << 16) | (unsigned)f2bf(sr);
    *(unsigned*)(sp + (size_t)c * kT) = pk;            // state ENTERING chunk c
    float ar = sl[(size_t)c * kT], ai = sl[(size_t)c * kT + 1];
    float t = lr * sr - li * si + ar;
    si = lr * si + li * sr + ai;
    sr = t;
  }
}

// ---------------- GEMM2: Y^T = U^T Mk^T + Spre^T P^T, GELU, -> y[h][m] ------
__global__ __launch_bounds__(256) void k_gemm2(const bf16_t* __restrict__ Xt,
                                               const bf16_t* __restrict__ SpreT,
                                               const bf16_t* __restrict__ Mkb,
                                               const bf16_t* __restrict__ Pb,
                                               bf16_t* __restrict__ Y) {
  int h = blockIdx.x;
  int tid = threadIdx.x, lane = tid & 31, w = tid >> 5;
  __shared__ __align__(16) bf16_t lds_mk[kT * kT];
  __shared__ __align__(16) bf16_t lds_p [kT * kT];
#if defined(ASYNC_LDS)
  for (int i = tid * 8; i < kT * kT; i += 256 * 8) {
    async_copy_b128(&Mkb[(size_t)h * (kT * kT) + i], &lds_mk[i]);
    async_copy_b128(&Pb [(size_t)h * (kT * kT) + i], &lds_p [i]);
  }
  wait_async0();
#else
  for (int i = tid * 8; i < kT * kT; i += 256 * 8) {
    *(uint4*)&lds_mk[i] = *(const uint4*)&Mkb[(size_t)h * (kT * kT) + i];
    *(uint4*)&lds_p [i] = *(const uint4*)&Pb [(size_t)h * (kT * kT) + i];
  }
#endif
  __syncthreads();

  int m0 = blockIdx.y * 128 + w * 16;
  int mp = m0 + (lane & 15);
  int kb = (lane >> 4) << 3, cb = (lane >> 4) << 4;
  const bf16_t* ub = Xt    + (size_t)h * kM + (mp >> 6) * kL + (mp & 63) * kT;
  const bf16_t* sb = SpreT + ((size_t)h * 512 + mp) * kT;
  v8f acc[4];
#pragma unroll
  for (int i = 0; i < 4; ++i) acc[i] = vzero8();
#pragma unroll
  for (int kk = 0; kk < kT; kk += 32) {
    BF16x16 au = load_seq16(ub + kk + cb);
    BF16x16 as = load_seq16(sb + kk + cb);
#pragma unroll
    for (int ct = 0; ct < 4; ++ct) {
      BF16x16 b1 = load_seq16(&lds_mk[(ct * 16 + (lane & 15)) * kT + kk + cb]);
      acc[ct] = wmma_bf16(au, b1, acc[ct]);
      BF16x16 b2 = load_seq16(&lds_p [(ct * 16 + (lane & 15)) * kT + kk + cb]);
      acc[ct] = wmma_bf16(as, b2, acc[ct]);
    }
  }
  int rb = m0 + 8 * (lane >> 4), cl = lane & 15;
#pragma unroll
  for (int ct = 0; ct < 4; ++ct)
#pragma unroll
    for (int r = 0; r < 8; ++r) {
      int mp2 = rb + r;
      float v = acc[ct][r];
      v = 0.5f * v * (1.f + erff(v * 0.70710678f));    // exact-erf GELU
      Y[(size_t)h * kM + (mp2 >> 6) * kL + (mp2 & 63) * kT + ct * 16 + cl] = f2bf(v);
    }
}

// ---------------- 1x1 conv GEMM + bias + GLU + residual ---------------------
// Double-buffered LDS staging of the 32(k) x 32(m) Y tile; async on CDNA5.
__global__ __launch_bounds__(256) void k_conv(const bf16_t* __restrict__ Wb,
                                              const float* __restrict__ bias,
                                              const bf16_t* __restrict__ Y,
                                              const float* __restrict__ Xin,
                                              float* __restrict__ V) {
  int m0 = blockIdx.x * 32;
  int tid = threadIdx.x, lane = tid & 31, w = tid >> 5;
  __shared__ __align__(16) bf16_t lds_y[2][32 * 32];
  v8f accA[8], accG[8];
#pragma unroll
  for (int i = 0; i < 8; ++i) { accA[i] = vzero8(); accG[i] = vzero8(); }
  int kb = (lane >> 4) << 3, kb16 = (lane >> 4) << 4;
  int oa = w * 64;                                      // this wave's a-rows

  auto stage = [&](int buf, int kk) {
    for (int i = tid; i < 512; i += 256) {              // 32x32 tile, 2 bf16/thread
      int k = i >> 4, mm = (i & 15) * 2;
#if defined(ASYNC_LDS)
      async_copy_b32(&Y[(size_t)(kk + k) * kM + m0 + mm], &lds_y[buf][k * 32 + mm]);
#else
      *(unsigned*)&lds_y[buf][k * 32 + mm] =
          *(const unsigned*)&Y[(size_t)(kk + k) * kM + m0 + mm];
#endif
    }
  };

  stage(0, 0);
#if defined(ASYNC_LDS)
  wait_async0();
#endif
  __syncthreads();

  int cur = 0;
  for (int kk = 0; kk < kH; kk += 32) {
    if (kk + 32 < kH) stage(cur ^ 1, kk + 32);          // overlap fetch with WMMA
    BF16x16 bfr[2];
#pragma unroll
    for (int ct = 0; ct < 2; ++ct) {
      BF16x16 f;
#pragma unroll
      for (int e = 0; e < 16; ++e)
        f.u[e] = lds_y[cur][(kb16 + e) * 32 + ct * 16 + (lane & 15)];
      bfr[ct] = f;
    }
#pragma unroll
    for (int rt = 0; rt < 4; ++rt) {
      BF16x16 aA = load_seq16(Wb + (size_t)(oa + rt * 16 + (lane & 15)) * kH + kk + kb16);
      BF16x16 aG = load_seq16(Wb + (size_t)(512 + oa + rt * 16 + (lane & 15)) * kH + kk + kb16);
#pragma unroll
      for (int ct = 0; ct < 2; ++ct) {
        accA[rt * 2 + ct] = wmma_bf16(aA, bfr[ct], accA[rt * 2 + ct]);
        accG[rt * 2 + ct] = wmma_bf16(aG, bfr[ct], accG[rt * 2 + ct]);
      }
    }
#if defined(ASYNC_LDS)
    wait_async0();
#endif
    __syncthreads();
    cur ^= 1;
  }
#pragma unroll
  for (int rt = 0; rt < 4; ++rt)
#pragma unroll
    for (int ct = 0; ct < 2; ++ct)
#pragma unroll
      for (int r = 0; r < 8; ++r) {
        int o = oa + rt * 16 + r + 8 * (lane >> 4);     // h index 0..511
        int m = m0 + ct * 16 + (lane & 15);
        float a = accA[rt * 2 + ct][r] + bias[o];
        float g = accG[rt * 2 + ct][r] + bias[512 + o];
        float rr = a * (1.f / (1.f + expf(-g)));        // GLU
        V[(size_t)m * kH + o] = rr + Xin[(size_t)m * kH + o];
      }
}

// ---------------- LayerNorm over h, write x[m][h] and Xt[h][m] --------------
__global__ __launch_bounds__(256) void k_ln(const float* __restrict__ V,
                                            const float* __restrict__ gamma,
                                            const float* __restrict__ beta,
                                            float* __restrict__ Xout,
                                            bf16_t* __restrict__ XtOut, int writeXt) {
  int m = blockIdx.x * 8 + (threadIdx.x >> 5);
  int lane = threadIdx.x & 31;
  const float* v = V + (size_t)m * kH;
  float vals[16], s = 0.f, s2 = 0.f;
#pragma unroll
  for (int i = 0; i < 16; ++i) {
    float x = v[i * 32 + lane];
    vals[i] = x; s += x; s2 += x * x;
  }
#pragma unroll
  for (int off = 16; off > 0; off >>= 1) {
    s += __shfl_xor(s, off, 32);
    s2 += __shfl_xor(s2, off, 32);
  }
  float mu = s * (1.f / kH);
  float var = s2 * (1.f / kH) - mu * mu;
  float rstd = rsqrtf(var + 1e-5f);
#pragma unroll
  for (int i = 0; i < 16; ++i) {
    int hh = i * 32 + lane;
    float o = (vals[i] - mu) * rstd * gamma[hh] + beta[hh];
    Xout[(size_t)m * kH + hh] = o;
    if (writeXt) XtOut[(size_t)hh * kM + m] = f2bf(o);
  }
}

// ---------------- host launcher ----------------
extern "C" void kernel_launch(void* const* d_in, const int* in_sizes, int n_in,
                              void* d_out, int out_size, void* d_ws, size_t ws_size,
                              hipStream_t stream) {
  const float* x_in    = (const float*)d_in[0];
  const float* log_dt  = (const float*)d_in[1];
  const float* A_imag  = (const float*)d_in[2];
  const float* logAre  = (const float*)d_in[3];
  const float* Cp      = (const float*)d_in[4];
  const float* Dp      = (const float*)d_in[5];
  const float* Wp      = (const float*)d_in[6];
  const float* bp      = (const float*)d_in[7];
  const float* gamma   = (const float*)d_in[8];
  const float* beta    = (const float*)d_in[9];

  char* p = (char*)d_ws;
  auto take = [&](size_t bytes) { char* r = p; p += (bytes + 255) & ~(size_t)255; return r; };
  bf16_t* Xt    = (bf16_t*)take((size_t)kH * kM * 2);
  bf16_t* Wb    = (bf16_t*)take((size_t)kNL * 2 * kH * kH * 2);
  bf16_t* Qb    = (bf16_t*)take((size_t)kH * kT * kT * 2);
  bf16_t* Mkb   = (bf16_t*)take((size_t)kH * kT * kT * 2);
  bf16_t* Pb    = (bf16_t*)take((size_t)kH * kT * kT * 2);
  float*  LamT  = (float* )take((size_t)kH * kN2 * 2 * 4);
  float*  SlocT = (float* )take((size_t)kH * 512 * kT * 4);
  bf16_t* SpreT = (bf16_t*)take((size_t)kH * 512 * kT * 2);
  bf16_t* Y     = (bf16_t*)take((size_t)kH * kM * 2);
  float*  V     = (float* )take((size_t)kM * kH * 4);
  float*  Xp0   = (float* )take((size_t)kM * kH * 4);
  float*  Xp1   = (float* )take((size_t)kM * kH * 4);
  float*  xbuf[2] = {Xp0, Xp1};

  k_tr<<<dim3(kB, kL / 32, kH / 32), dim3(32, 8), 0, stream>>>(x_in, Xt);
  {
    int n = kNL * 2 * kH * kH;
    k_wcvt<<<(n + 255) / 256, 256, 0, stream>>>(Wp, Wb, n);
  }
  for (int lay = 0; lay < kNL; ++lay) {
    k_gen<<<kH, 32, 0, stream>>>(lay, log_dt, A_imag, logAre, Cp, Dp, Qb, Mkb, Pb, LamT);
    k_gemm1<<<dim3(kH, 4), 256, 0, stream>>>(Xt, Qb, SlocT);
    k_cscan<<<kH * kB, 32, 0, stream>>>(LamT, SlocT, SpreT);
    k_gemm2<<<dim3(kH, 4), 256, 0, stream>>>(Xt, SpreT, Mkb, Pb, Y);
    const float* xres = (lay == 0) ? x_in : xbuf[(lay - 1) & 1];
    k_conv<<<kM / 32, 256, 0, stream>>>(Wb + (size_t)lay * 2 * kH * kH,
                                        bp + lay * 2 * kH, Y, xres, V);
    float* xo = (lay == kNL - 1) ? (float*)d_out : xbuf[lay & 1];
    k_ln<<<kM / 8, 256, 0, stream>>>(V, gamma + lay * kH, beta + lay * kH,
                                     xo, Xt, (lay != kNL - 1) ? 1 : 0);
  }
  (void)in_sizes; (void)n_in; (void)out_size; (void)ws_size;
}
